// LSTMDecoderModule_18597208392293
// MI455X (gfx1250) — matrix operand order repaired and checked by reading, not compiled, not run on hardware
//
#include <hip/hip_runtime.h>
#include <hip/hip_bf16.h>

// ---------------- problem constants ----------------
#define TMAX   1024
#define RINGN  8          // ring slots per inter-step buffer (power of 2)
#define NWGL   16         // workgroups per LSTM layer
#define BATCH  64
#define HDIM   1024
#define GDIM   4096       // 4*HDIM gates
#define EDIM   1024
#define ODIM   64
#define NLAYER 3

typedef __attribute__((ext_vector_type(16))) __bf16 v16bf;
typedef __attribute__((ext_vector_type(8)))  __bf16 v8bf;
typedef __attribute__((ext_vector_type(8)))  float  v8f;
typedef __attribute__((ext_vector_type(4)))  unsigned int u32x4;

// LDS (address_space(3)) typed access so A-fragment reads lower to ds_load_b128
typedef __attribute__((address_space(3))) const u32x4 cu32x4_lds;
// Global (address_space(1)) typed access so laundered weight streams stay global_load_b128
typedef __attribute__((address_space(1))) const __bf16 gcbf16;
typedef __attribute__((address_space(1))) const u32x4 gcu32x4;

// ---------------- workspace layout (bytes) ----------------
#define OFF_WHH   0ull
#define SZ_WHH    (3ull * GDIM * HDIM * 2)     // bf16 W_hh, 24 MB
#define OFF_WIH   (OFF_WHH + SZ_WHH)
#define SZ_WIH    (2ull * GDIM * EDIM * 2)     // bf16 W_ih layers 1,2, 16 MB
#define OFF_WLIN  (OFF_WIH + SZ_WIH)
#define SZ_WLIN   ((size_t)ODIM * HDIM * 2)    // bf16 W_lin
#define OFF_BIAS  (OFF_WLIN + SZ_WLIN)
#define SZ_BIAS   (3ull * GDIM * 4)            // fp32 b_ih+b_hh
#define OFF_XP0   (OFF_BIAS + SZ_BIAS)
#define SZ_XP0    ((size_t)BATCH * GDIM * 4)   // fp32 layer-0 input proj (const over t)
#define OFF_RING  (OFF_XP0 + SZ_XP0)
#define SZ_RING   (3ull * RINGN * BATCH * HDIM * 2) // bf16 h rings, 3 MB
#define OFF_DONE  (OFF_RING + SZ_RING)
#define OFF_FREED (OFF_DONE + 16384ull)

__device__ __forceinline__ int ring_slot(int t) { return (t + RINGN) & (RINGN - 1); }
__device__ __forceinline__ float fsig(float x) { return 1.0f / (1.0f + __expf(-x)); }

// launder a pointer so loads through it cannot be hoisted/CSE'd across timesteps,
// rebuilding it as an address_space(1) pointer so codegen keeps global_load form.
#define LAUNDER_GPTR(p) ({ size_t x_ = (size_t)(p); asm volatile("" : "+v"(x_)); (gcbf16*)x_; })

// CDNA5 async copy: 16B global -> LDS, tracked by ASYNCcnt
__device__ __forceinline__ void async_copy16(unsigned lds_off, const void* gptr) {
    asm volatile("global_load_async_to_lds_b128 %0, %1, off"
                 :: "v"(lds_off), "v"(gptr) : "memory");
}
__device__ __forceinline__ void wait_async0() {
    asm volatile("s_wait_asynccnt 0x0" ::: "memory");
}

// build a 16-half lane fragment out of two 16-byte chunks (generic)
__device__ __forceinline__ v16bf ld_frag2(const __bf16* p0, const __bf16* p1) {
    union U { u32x4 u; v8bf b; };
    U x, y;
    x.u = *reinterpret_cast<const u32x4*>(p0);
    y.u = *reinterpret_cast<const u32x4*>(p1);
    return __builtin_shufflevector(x.b, y.b, 0,1,2,3,4,5,6,7,8,9,10,11,12,13,14,15);
}

// same, from explicit global address space
__device__ __forceinline__ v16bf ld_frag2_g(gcbf16* p0, gcbf16* p1) {
    union U { u32x4 u; v8bf b; };
    U x, y;
    x.u = *(gcu32x4*)p0;
    y.u = *(gcu32x4*)p1;
    return __builtin_shufflevector(x.b, y.b, 0,1,2,3,4,5,6,7,8,9,10,11,12,13,14,15);
}

// same, from LDS via 32-bit LDS byte offsets -> ds_load_b128
__device__ __forceinline__ v16bf ld_frag2_lds(unsigned o0, unsigned o1) {
    union U { u32x4 u; v8bf b; };
    U x, y;
    x.u = *(cu32x4_lds*)o0;
    y.u = *(cu32x4_lds*)o1;
    return __builtin_shufflevector(x.b, y.b, 0,1,2,3,4,5,6,7,8,9,10,11,12,13,14,15);
}

// A (16x32 bf16, M=batch rows, row-major src [rows][1024]):
// lane<16: row m0+lane,  K = K0+{0..7,16..23}; lane>=16: row m0+lane-16, K = K0+{8..15,24..31}
__device__ __forceinline__ v16bf ldA(const __bf16* mat, int m0, int K0, int lane) {
    int row = m0 + (lane & 15);
    int k0  = K0 + ((lane >> 4) << 3);
    const __bf16* p = mat + (size_t)row * HDIM + k0;
    return ld_frag2(p, p + 16);
}

// A fragment from LDS-staged h (lbase = LDS byte offset of the staging buffer)
__device__ __forceinline__ v16bf ldA_lds(unsigned lbase, int m0, int K0, int lane) {
    int row = m0 + (lane & 15);
    int k0  = K0 + ((lane >> 4) << 3);
    unsigned off = lbase + (unsigned)(row * HDIM + k0) * 2u;
    return ld_frag2_lds(off, off + 32u);
}

// B (32x16 bf16, N=gate/out col, K contiguous in row-major W[g][k]):
// lane<16: col g0+lane holds K0+{0..7,16..23}; lane>=16: col g0+lane-16 holds K0+{8..15,24..31}
__device__ __forceinline__ v16bf ldB_g(gcbf16* W, int g0, int K0, int lane) {
    int g  = g0 + (lane & 15);
    int k0 = K0 + ((lane >> 4) << 3);
    gcbf16* p = W + (size_t)g * HDIM + k0;
    return ld_frag2_g(p, p + 16);
}

__device__ __forceinline__ void wait_ge(const unsigned* p, unsigned target) {
    while (__hip_atomic_load(p, __ATOMIC_ACQUIRE, __HIP_MEMORY_SCOPE_AGENT) < target) {
        __builtin_amdgcn_s_sleep(1);
    }
}
__device__ __forceinline__ void sig_add(unsigned* p, unsigned v) {
    __hip_atomic_fetch_add(p, v, __ATOMIC_RELEASE, __HIP_MEMORY_SCOPE_AGENT);
}

// ---------------- kernel 1: convert weights to bf16, init rings/counters ----------------
__global__ void lstm_init_kernel(const float* __restrict__ Wih, const float* __restrict__ Whh,
                                 const float* __restrict__ bih, const float* __restrict__ bhh,
                                 const float* __restrict__ Wlin, const float* __restrict__ h0,
                                 __bf16* __restrict__ whh_bf, __bf16* __restrict__ wih_bf,
                                 __bf16* __restrict__ wlin_bf, float* __restrict__ bias,
                                 __bf16* __restrict__ ring,
                                 unsigned* __restrict__ done, unsigned* __restrict__ freed) {
    size_t i = (size_t)blockIdx.x * blockDim.x + threadIdx.x;
    size_t stride = (size_t)gridDim.x * blockDim.x;
    for (size_t e = i; e < 3ull * GDIM * HDIM; e += stride) whh_bf[e] = (__bf16)Whh[e];
    for (size_t e = i; e < 2ull * GDIM * EDIM; e += stride)
        wih_bf[e] = (__bf16)Wih[(size_t)GDIM * EDIM + e];           // layers 1,2
    for (size_t e = i; e < (size_t)ODIM * HDIM; e += stride) wlin_bf[e] = (__bf16)Wlin[e];
    for (size_t e = i; e < 3ull * GDIM; e += stride) bias[e] = bih[e] + bhh[e];
    // h0 -> ring slot RINGN-1 (t = -1) per layer
    for (size_t e = i; e < 3ull * BATCH * HDIM; e += stride) {
        size_t l = e >> 16, rem = e & 65535;
        ring[(l * RINGN + (RINGN - 1)) * (size_t)(BATCH * HDIM) + rem] = (__bf16)h0[e];
    }
    // counters: done[l][0]=16 (h0 produced); freed[l][0] preset so slot -1 completes
    for (size_t e = i; e < 3ull * (TMAX + 1); e += stride) {
        unsigned li = (unsigned)(e / (TMAX + 1));
        unsigned r  = (unsigned)(e % (TMAX + 1));
        done[e]  = (r == 0) ? 16u : 0u;
        freed[e] = (r == 0) ? ((li == 2) ? 1u : 16u) : 0u;
    }
}

// ---------------- kernel 2: layer-0 input projection (constant over t) ----------------
__global__ void lstm_xproj0_kernel(const float* __restrict__ enc, const float* __restrict__ Wih0,
                                   const float* __restrict__ bih, const float* __restrict__ bhh,
                                   float* __restrict__ xp0) {
    int idx = blockIdx.x * 256 + threadIdx.x;   // 64*4096 outputs
    int b = idx >> 12, g = idx & (GDIM - 1);
    const float4* er = (const float4*)(enc + (size_t)b * EDIM);
    const float4* wr = (const float4*)(Wih0 + (size_t)g * EDIM);
    float acc = bih[g] + bhh[g];
    #pragma unroll 8
    for (int k = 0; k < EDIM / 4; ++k) {
        float4 a = er[k], w = wr[k];
        acc += a.x * w.x + a.y * w.y + a.z * w.z + a.w * w.w;
    }
    xp0[(size_t)b * GDIM + g] = acc;
}

// ---------------- kernel 3: persistent pipelined LSTM + output GEMM ----------------
__global__ __launch_bounds__(256, 1)
void lstm_pipeline_kernel(const float* __restrict__ c0, const float* __restrict__ xp0,
                          const __bf16* __restrict__ whh, const __bf16* __restrict__ wih,
                          const __bf16* __restrict__ wlin, const float* __restrict__ bias,
                          const float* __restrict__ blin,
                          __bf16* __restrict__ ring,
                          unsigned* __restrict__ done, unsigned* __restrict__ freed,
                          float* __restrict__ out, const int* __restrict__ Tp) {
    __shared__ __align__(16) __bf16 lds_h[BATCH * HDIM];   // 128 KB: h_{t-1} staged per step

    const int tid  = threadIdx.x;
    const int lane = tid & 31;
    const int wv   = tid >> 5;          // wave id 0..7 (wave32)
    const int T    = *Tp;

    if (blockIdx.x < NLAYER * NWGL) {
        // ------------- LSTM layer stage -------------
        // Wave wv owns: gate chunks i,f,g,o for 16 hidden cols (j16 = wv&3) and
        // M-tile pair (mh = wv>>2). So the full cell update is lane-local.
        const int l   = blockIdx.x >> 4;
        const int w   = blockIdx.x & 15;
        const int j16 = wv & 3;
        const int mh  = wv >> 2;
        const int hi8 = (lane >> 4) << 3;
        const int jcol = w * 64 + j16 * 16 + (lane & 15);   // hidden column this lane owns
        const unsigned consl = (l == 2) ? 17u : 32u;
        const unsigned lbase = (unsigned)(size_t)(void*)lds_h;  // LDS byte offset (addr[31:0])

        // cell state lives in registers for the whole sequence
        float cr[2][8];
        #pragma unroll
        for (int mi = 0; mi < 2; ++mi)
            #pragma unroll
            for (int r = 0; r < 8; ++r) {
                int b = (2 * mh + mi) * 16 + r + hi8;
                cr[mi][r] = c0[((size_t)l * BATCH + b) * HDIM + jcol];
            }

        for (int t = 0; t < T; ++t) {
            if (tid == 0) {
                wait_ge(&done[l * (TMAX + 1) + t], 16u);                       // h_{t-1} ready
                if (l > 0) wait_ge(&done[(l - 1) * (TMAX + 1) + t + 1], 16u);  // x_t ready
                if (t >= RINGN) wait_ge(&freed[l * (TMAX + 1) + (t - RINGN + 1)], consl);
            }
            __syncthreads();

            // async-stage h_{t-1} (128 KB) into LDS; ASYNCcnt tracked
            const __bf16* hprev = ring + ((size_t)l * RINGN + ring_slot(t - 1)) * (BATCH * HDIM);
            for (int i = tid; i < (BATCH * HDIM) / 8; i += 256)
                async_copy16(lbase + i * 16, (const char*)hprev + (size_t)i * 16);
            wait_async0();
            __syncthreads();

            // accumulators: acc[mi][ci] = gate chunk ci (i,f,g,o), N-tile = ci*4 + j16
            v8f acc[2][4];
            #pragma unroll
            for (int mi = 0; mi < 2; ++mi)
                #pragma unroll
                for (int ci = 0; ci < 4; ++ci) {
                    int gc = ci * HDIM + jcol;
                    if (l == 0) {
                        #pragma unroll
                        for (int r = 0; r < 8; ++r)
                            acc[mi][ci][r] =
                                xp0[(size_t)((2 * mh + mi) * 16 + r + hi8) * GDIM + gc];
                    } else {
                        float bv = bias[l * GDIM + gc];
                        #pragma unroll
                        for (int r = 0; r < 8; ++r) acc[mi][ci][r] = bv;
                    }
                }

            // recurrent GEMM: A from LDS (ds_load_b128), B = W_hh global stream (laundered)
            {
                gcbf16* Wm = LAUNDER_GPTR(whh + (size_t)l * GDIM * HDIM);
                for (int K0 = 0; K0 < HDIM; K0 += 32) {
                    v16bf af[2];
                    #pragma unroll
                    for (int mi = 0; mi < 2; ++mi)
                        af[mi] = ldA_lds(lbase, (2 * mh + mi) * 16, K0, lane);
                    #pragma unroll
                    for (int ci = 0; ci < 4; ++ci) {
                        int g0 = ci * HDIM + w * 64 + j16 * 16;
                        v16bf bf = ldB_g(Wm, g0, K0, lane);
                        #pragma unroll
                        for (int mi = 0; mi < 2; ++mi)
                            acc[mi][ci] = __builtin_amdgcn_wmma_f32_16x16x32_bf16(
                                false, af[mi], false, bf, (short)0, acc[mi][ci], false, false);
                    }
                }
            }
            // input GEMM (layers 1,2): A = x_t from previous layer's ring (global)
            if (l > 0) {
                const __bf16* Ax =
                    ring + ((size_t)(l - 1) * RINGN + ring_slot(t)) * (BATCH * HDIM);
                gcbf16* Wm = LAUNDER_GPTR(wih + (size_t)(l - 1) * GDIM * EDIM);
                for (int K0 = 0; K0 < HDIM; K0 += 32) {
                    v16bf af[2];
                    #pragma unroll
                    for (int mi = 0; mi < 2; ++mi) af[mi] = ldA(Ax, (2 * mh + mi) * 16, K0, lane);
                    #pragma unroll
                    for (int ci = 0; ci < 4; ++ci) {
                        int g0 = ci * HDIM + w * 64 + j16 * 16;
                        v16bf bf = ldB_g(Wm, g0, K0, lane);
                        #pragma unroll
                        for (int mi = 0; mi < 2; ++mi)
                            acc[mi][ci] = __builtin_amdgcn_wmma_f32_16x16x32_bf16(
                                false, af[mi], false, bf, (short)0, acc[mi][ci], false, false);
                    }
                }
            }

            // lane-local LSTM cell update; emit h_t as bf16 straight from registers
            __bf16* hout = ring + ((size_t)l * RINGN + ring_slot(t)) * (BATCH * HDIM);
            #pragma unroll
            for (int mi = 0; mi < 2; ++mi)
                #pragma unroll
                for (int r = 0; r < 8; ++r) {
                    float iv = acc[mi][0][r], fv = acc[mi][1][r];
                    float gv = acc[mi][2][r], ov = acc[mi][3][r];
                    float c = fsig(fv) * cr[mi][r] + fsig(iv) * tanhf(gv);
                    cr[mi][r] = c;
                    float h = fsig(ov) * tanhf(c);
                    int b = (2 * mh + mi) * 16 + r + hi8;
                    hout[(size_t)b * HDIM + jcol] = (__bf16)h;
                }
            __threadfence();
            __syncthreads();
            if (tid == 0) {
                sig_add(&done[l * (TMAX + 1) + t + 1], 1u);          // h_t produced
                sig_add(&freed[l * (TMAX + 1) + t], 1u);             // consumed own slot t-1
                if (l > 0) sig_add(&freed[(l - 1) * (TMAX + 1) + t + 1], 1u); // consumed input t
            }
        }
    } else {
        // ------------- output projection stage: out[b][t][o] = h2_t @ W_lin^T + b_lin -------------
        const int ow = blockIdx.x - NLAYER * NWGL;   // 0..7
        for (int t = ow; t < T; t += 8) {
            if (tid == 0) wait_ge(&done[2 * (TMAX + 1) + t + 1], 16u);
            __syncthreads();
            const __bf16* hm = ring + ((size_t)2 * RINGN + ring_slot(t)) * (BATCH * HDIM);
            gcbf16* wl = LAUNDER_GPTR(wlin);
            int mt = wv >> 1;                // 4 M-tiles over 8 waves, 2 N-tiles each
            v8f acc[2];
            #pragma unroll
            for (int q = 0; q < 2; ++q) {
                int nt = ((wv & 1) << 1) + q;
                float bv = blin[nt * 16 + (lane & 15)];
                #pragma unroll
                for (int r = 0; r < 8; ++r) acc[q][r] = bv;
            }
            for (int K0 = 0; K0 < HDIM; K0 += 32) {
                v16bf a = ldA(hm, mt * 16, K0, lane);
                #pragma unroll
                for (int q = 0; q < 2; ++q) {
                    int nt = ((wv & 1) << 1) + q;
                    v16bf b = ldB_g(wl, nt * 16, K0, lane);
                    acc[q] = __builtin_amdgcn_wmma_f32_16x16x32_bf16(
                        false, a, false, b, (short)0, acc[q], false, false);
                }
            }
            int hi8 = (lane >> 4) << 3;
            #pragma unroll
            for (int q = 0; q < 2; ++q) {
                int nt = ((wv & 1) << 1) + q;
                int o = nt * 16 + (lane & 15);
                #pragma unroll
                for (int r = 0; r < 8; ++r) {
                    int b = mt * 16 + r + hi8;
                    out[((size_t)b * T + t) * ODIM + o] = acc[q][r];
                }
            }
            __syncthreads();
            if (tid == 0) sig_add(&freed[2 * (TMAX + 1) + t + 1], 1u);
        }
    }
}

// ---------------- host side ----------------
extern "C" void kernel_launch(void* const* d_in, const int* in_sizes, int n_in,
                              void* d_out, int out_size, void* d_ws, size_t ws_size,
                              hipStream_t stream) {
    (void)in_sizes; (void)n_in; (void)out_size; (void)ws_size;
    const float* enc  = (const float*)d_in[0];
    const float* h0   = (const float*)d_in[1];
    const float* c0   = (const float*)d_in[2];
    const float* Wih  = (const float*)d_in[3];
    const float* Whh  = (const float*)d_in[4];
    const float* bih  = (const float*)d_in[5];
    const float* bhh  = (const float*)d_in[6];
    const float* Wlin = (const float*)d_in[7];
    const float* blin = (const float*)d_in[8];
    const int*   Tp   = (const int*)d_in[9];
    float* out = (float*)d_out;

    char* ws = (char*)d_ws;
    __bf16*  whh_bf  = (__bf16*)(ws + OFF_WHH);
    __bf16*  wih_bf  = (__bf16*)(ws + OFF_WIH);
    __bf16*  wlin_bf = (__bf16*)(ws + OFF_WLIN);
    float*   bias    = (float*)(ws + OFF_BIAS);
    float*   xp0     = (float*)(ws + OFF_XP0);
    __bf16*  ring    = (__bf16*)(ws + OFF_RING);
    unsigned* done   = (unsigned*)(ws + OFF_DONE);
    unsigned* freed  = (unsigned*)(ws + OFF_FREED);

    lstm_init_kernel<<<dim3(2048), dim3(256), 0, stream>>>(
        Wih, Whh, bih, bhh, Wlin, h0, whh_bf, wih_bf, wlin_bf, bias, ring, done, freed);

    lstm_xproj0_kernel<<<dim3((BATCH * GDIM) / 256), dim3(256), 0, stream>>>(
        enc, Wih, bih, bhh, xp0);

    lstm_pipeline_kernel<<<dim3(NLAYER * NWGL + 8), dim3(256), 0, stream>>>(
        c0, xp0, whh_bf, wih_bf, wlin_bf, bias, blin, ring, done, freed, out, Tp);
}
